// Critic_32401233281456
// MI455X (gfx1250) — matrix operand order repaired and checked by reading, not compiled, not run on hardware
//
#include <hip/hip_runtime.h>

// ---------------- problem constants ----------------
#define N_AGENTS 30
#define HID      512
#define INF      14
#define BATCH    8192
#define MT       64          // batch rows per workgroup

// ---------------- LDS layout (bytes) ----------------
#define ASTR 520             // halves per A row (4-dword bank stagger)
#define YSTR 516             // floats per Y row
#define BSTR 520             // halves per BT row (TDM pad: 256dw + 4dw)
#define A_OFF    0
#define A_BYTES  (MT * ASTR * 2)                 // 66560
#define Y_OFF    (A_OFF + A_BYTES)
#define Y_BYTES  (MT * YSTR * 4)                 // 132096
#define BT_OFF   (Y_OFF + Y_BYTES)               // 198656  (dedicated: TDM target)
#define BT_BYTES (MT * BSTR * 2)                 // 66560
#define ST_OFF   (BT_OFF + BT_BYTES)             // 265216
#define ST_BYTES ((64 + 64 + 512) * 4)           // MU/RS/RED = 2560
#define W1C_OFF  (ST_OFF + ST_BYTES)             // 267776 (dedicated, no overlay)
#define S1_OFF   (W1C_OFF + HID * 16 * 4)        // 300544
#define SMEM_BYTES (S1_OFF + MT * 16 * 4)        // 304640  (<= 320KB WGP LDS)

typedef __attribute__((ext_vector_type(16))) _Float16 v16h;
typedef __attribute__((ext_vector_type(8)))  _Float16 v8h;
typedef __attribute__((ext_vector_type(8)))  float    v8f;
typedef __attribute__((ext_vector_type(2)))  float    v2f;
typedef __attribute__((ext_vector_type(4)))  unsigned uint32x4;
typedef __attribute__((ext_vector_type(8)))  int      int32x8;
typedef __attribute__((ext_vector_type(4)))  int      int32x4;

// ---------------- prep: w2 f32 -> f16 in workspace ----------------
__global__ __launch_bounds__(256)
void w2_cvt_kernel(const float* __restrict__ w2, _Float16* __restrict__ w2h) {
    int i = blockIdx.x * 256 + threadIdx.x;
    w2h[i] = (_Float16)w2[i];
}

#if __has_builtin(__builtin_amdgcn_tensor_load_to_lds)
// Issue one TDM descriptor: 2D tile 512(k) x 64(n) of f16 from gsrc into LDS at
// ldst, padding +4 dwords every 256 dwords -> LDS row stride 520 halves.
// (D# bit layout per CDNA5 ISA.)
__device__ __forceinline__ void tdm_load_tile(const _Float16* gsrc, _Float16* ldst) {
    unsigned ldsoff = (unsigned)(unsigned long long)ldst;  // low32 of flat = LDS offset
    unsigned long long ga = (unsigned long long)(const void*)gsrc;
    uint32x4 d0;
    d0[0] = 1u;                                            // count=1, user mode
    d0[1] = ldsoff;                                        // lds_addr
    d0[2] = (unsigned)(ga & 0xffffffffull);                // global_addr lo
    d0[3] = (unsigned)((ga >> 32) & 0x1ffffffull) | (2u << 30); // addr hi | type=2
    int32x8 d1;
    d1[0] = (1 << 16) | (1 << 20) | (7 << 22) | (3 << 25); // 2B elems, pad 256dw->+4dw
    d1[1] = (int)(512u << 16);   // tensor_dim0 = 512 (lo16 at bits 63:48)
    d1[2] = (int)(512u << 16);   // tensor_dim1 = 512
    d1[3] = (int)(512u << 16);   // tile_dim0   = 512
    d1[4] = 64;                  // tile_dim1 = 64, tile_dim2 = 0
    d1[5] = 512;                 // tensor_dim0_stride = 512
    d1[6] = 0;                   // stride0 hi | stride1 lo16
    d1[7] = 4;                   // tensor_dim1_stride = 262144 (bits 47:16)
    int32x4 d2 = {0, 0, 0, 0};
    int32x4 d3 = {0, 0, 0, 0};
    int32x8 dx = {0, 0, 0, 0, 0, 0, 0, 0};
    __builtin_amdgcn_tensor_load_to_lds(d0, d1, d2, d3, dx, 0);
}
#endif

// ---------------- fused critic ----------------
__global__ __launch_bounds__(256)
void critic_fused_kernel(const float* __restrict__ inputs,
                         const float* __restrict__ actions,
                         const float* __restrict__ w1,
                         const float* __restrict__ b1,
                         const float* __restrict__ g1,
                         const float* __restrict__ be1,
                         const float* __restrict__ b2,
                         const float* __restrict__ g2,
                         const float* __restrict__ be2,
                         const float* __restrict__ wV,
                         const float* __restrict__ bV,
                         const _Float16* __restrict__ w2h,
                         float* __restrict__ out)
{
    extern __shared__ char smem[];
    _Float16* As  = (_Float16*)(smem + A_OFF);    // s2, f16, [MT][ASTR]
    float*    Yb  = (float*)   (smem + Y_OFF);    // y1 then y2, [MT][YSTR]
    _Float16* BT  = (_Float16*)(smem + BT_OFF);   // w2 tile, [64][BSTR]
    float*    MU  = (float*)   (smem + ST_OFF);
    float*    RS  = MU + 64;
    float*    RED = RS + 64;                      // 512 floats
    float*    W1C = (float*)   (smem + W1C_OFF);  // w1 padded, [HID][16]
    float*    S1  = (float*)   (smem + S1_OFF);   // [MT][16]

    const int tid = threadIdx.x;
    const int wv  = tid >> 5;
    const int l   = tid & 31;
    const int lm  = l & 15;      // lane row/col within 16
    const int lg  = l >> 4;      // lane group
    const int b0  = blockIdx.x * MT;

#if __has_builtin(__builtin_amdgcn_tensor_load_to_lds)
    // Prefetch first w2 tile via TDM; overlaps all of stages 0..1.75.
    if (wv == 0) tdm_load_tile(w2h, BT);
#endif

    // ---- Stage 0: stage padded w1 (as [n][k16]) and agent-summed inputs s1 ----
    for (int i = tid; i < HID * 16; i += 256) {
        int n = i >> 4, k = i & 15;
        W1C[i] = (k < INF) ? w1[n * INF + k] : 0.0f;
    }
    for (int i = tid; i < MT * 16; i += 256) {
        int r = i >> 4, f = i & 15;
        float s = 0.0f;
        if (f < INF) {
            const int b = b0 + r;
            #pragma unroll
            for (int a = 0; a < N_AGENTS; ++a) {
                int c = a * INF + f;
                s += (c < 12 * N_AGENTS) ? inputs[b * (12 * N_AGENTS) + c]
                                         : actions[b * (2 * N_AGENTS) + (c - 12 * N_AGENTS)];
            }
        }
        S1[i] = s;
    }
    __syncthreads();

    // ---- Stage 1: y1 = s1 @ w1^T + b1  (exact f32 WMMA, K=14 padded to 16) ----
    for (int t = wv * 16; t < wv * 16 + 16; ++t) {
        int rt = t >> 5, ct = t & 31;
        int col = ct * 16 + lm;
        float bias = b1[col];
        v8f acc;
        #pragma unroll
        for (int i = 0; i < 8; ++i) acc[i] = bias;
        const float* s1row = &S1[(rt * 16 + lm) * 16];
        const float* wrow  = &W1C[col * 16];
        #pragma unroll
        for (int j = 0; j < 4; ++j) {
            // A 16x4 f32: a[v] = A[m][4j + 2g + v];  B 4x16: b[v] = B[4j+2g+v][n]
            v2f af = *(const v2f*)&s1row[j * 4 + 2 * lg];
            v2f bf = *(const v2f*)&wrow [j * 4 + 2 * lg];
            acc = __builtin_amdgcn_wmma_f32_16x16x4_f32(false, af, false, bf,
                                                        (short)0, acc, false, false);
        }
        int row0 = rt * 16 + 8 * lg;
        #pragma unroll
        for (int i = 0; i < 8; ++i) Yb[(row0 + i) * YSTR + col] = acc[i];
    }
    __syncthreads();

    // ---- Stage 1.5: LN stats of y1 (4 threads per row) ----
    {
        int r = tid >> 2, q = tid & 3;
        const float* yr = &Yb[r * YSTR];
        float s = 0.f, ss = 0.f;
        for (int i = 0; i < 128; ++i) { float v = yr[i * 4 + q]; s += v; ss += v * v; }
        RED[tid * 2] = s; RED[tid * 2 + 1] = ss;
        __syncthreads();
        if (q == 0) {
            float S = 0.f, SS = 0.f;
            #pragma unroll
            for (int i = 0; i < 4; ++i) { S += RED[(tid + i) * 2]; SS += RED[(tid + i) * 2 + 1]; }
            float mu  = S * (1.0f / (float)HID);
            float var = SS * (1.0f / (float)HID) - mu * mu;
            MU[r] = mu; RS[r] = rsqrtf(var + 1e-5f);
        }
        __syncthreads();
    }

    // ---- Stage 1.75: s2[h] = sum_a relu(n1*g1 + be1)  -> As (f16 A-matrix) ----
    for (int h = tid; h < HID; h += 256) {
        #pragma unroll 1
        for (int rb = 0; rb < MT; rb += 16) {
            float nrm[16], acc[16];
            #pragma unroll
            for (int r = 0; r < 16; ++r) {
                nrm[r] = (Yb[(rb + r) * YSTR + h] - MU[rb + r]) * RS[rb + r];
                acc[r] = 0.f;
            }
            for (int a = 0; a < N_AGENTS; ++a) {
                float g  = g1 [a * HID + h];
                float bb = be1[a * HID + h];
                #pragma unroll
                for (int r = 0; r < 16; ++r) {
                    float v = fmaf(nrm[r], g, bb);
                    acc[r] += fmaxf(v, 0.f);
                }
            }
            #pragma unroll
            for (int r = 0; r < 16; ++r)
                As[(rb + r) * ASTR + h] = (_Float16)acc[r];
        }
    }
    __syncthreads();

    // ---- Stage 2: y2 = s2 @ w2^T + b2  (f16 WMMA, TDM-staged w2 tiles) ----
    const int rt2 = wv & 3;          // row tile (16 rows)
    const int ch  = wv >> 2;         // column half within 64-col tile
    const _Float16* arow = &As[(rt2 * 16 + lm) * ASTR];

    // A fragments register-resident across all 8 column passes (128 VGPRs;
    // 1 WG/WGP -> 2 waves/SIMD -> ample budget, halves hot-loop LDS traffic).
    v16h afrag[16];
    #pragma unroll
    for (int kc = 0; kc < 16; ++kc) {
        v8h alo = *(const v8h*)(arow + kc * 32 + 8 * lg);
        v8h ahi = *(const v8h*)(arow + kc * 32 + 16 + 8 * lg);
        #pragma unroll
        for (int i = 0; i < 8; ++i) { afrag[kc][i] = alo[i]; afrag[kc][8 + i] = ahi[i]; }
    }

    #pragma unroll 1
    for (int ns = 0; ns < 8; ++ns) {
        const int n0 = ns * 64;
#if __has_builtin(__builtin_amdgcn_tensor_load_to_lds)
        if (wv == 0) __builtin_amdgcn_s_wait_tensorcnt(0);   // tile ns landed
        __syncthreads();
#else
        for (int i = tid; i < 64 * HID; i += 256) {
            int n = i >> 9, k = i & (HID - 1);
            BT[n * BSTR + k] = w2h[(unsigned long long)(n0 + n) * HID + k];
        }
        __syncthreads();
#endif

        const int colA = n0 + ch * 32 + lm;
        float bias0 = b2[colA], bias1 = b2[colA + 16];
        v8f acc0, acc1;
        #pragma unroll
        for (int i = 0; i < 8; ++i) { acc0[i] = bias0; acc1[i] = bias1; }
        const _Float16* brow0 = &BT[(ch * 32 + lm) * BSTR];
        const _Float16* brow1 = &BT[(ch * 32 + 16 + lm) * BSTR];
        #pragma unroll
        for (int kc = 0; kc < 16; ++kc) {
            const int kb = kc * 32;
            // B frag: halves k in [8g,8g+8) and [16+8g,24+8g) per column (2x ds_load_b128)
            v8h b0l = *(const v8h*)(brow0 + kb + 8 * lg);
            v8h b0h = *(const v8h*)(brow0 + kb + 16 + 8 * lg);
            v16h bb0;
            #pragma unroll
            for (int i = 0; i < 8; ++i) { bb0[i] = b0l[i]; bb0[8 + i] = b0h[i]; }
            acc0 = __builtin_amdgcn_wmma_f32_16x16x32_f16(false, afrag[kc], false, bb0,
                                                          (short)0, acc0, false, false);
            v8h b1l = *(const v8h*)(brow1 + kb + 8 * lg);
            v8h b1h = *(const v8h*)(brow1 + kb + 16 + 8 * lg);
            v16h bb1;
            #pragma unroll
            for (int i = 0; i < 8; ++i) { bb1[i] = b1l[i]; bb1[8 + i] = b1h[i]; }
            acc1 = __builtin_amdgcn_wmma_f32_16x16x32_f16(false, afrag[kc], false, bb1,
                                                          (short)0, acc1, false, false);
        }
        const int row0 = rt2 * 16 + 8 * lg;
        #pragma unroll
        for (int i = 0; i < 8; ++i) {
            Yb[(row0 + i) * YSTR + colA]      = acc0[i];
            Yb[(row0 + i) * YSTR + colA + 16] = acc1[i];
        }
        __syncthreads();             // all reads of tile ns done -> safe to refill
#if __has_builtin(__builtin_amdgcn_tensor_load_to_lds)
        if (wv == 0 && ns + 1 < 8)
            tdm_load_tile(w2h + (unsigned long long)(n0 + 64) * HID, BT);
#endif
    }
    __syncthreads();

    // ---- Stage 2.5: LN stats of y2, agent sum with g2/be2, dot with wV ----
    {
        int r = tid >> 2, q = tid & 3;
        const float* yr = &Yb[r * YSTR];
        float s = 0.f, ss = 0.f;
        for (int i = 0; i < 128; ++i) { float v = yr[i * 4 + q]; s += v; ss += v * v; }
        RED[tid * 2] = s; RED[tid * 2 + 1] = ss;
        __syncthreads();
        if (q == 0) {
            float S = 0.f, SS = 0.f;
            #pragma unroll
            for (int i = 0; i < 4; ++i) { S += RED[(tid + i) * 2]; SS += RED[(tid + i) * 2 + 1]; }
            float mu  = S * (1.0f / (float)HID);
            float var = SS * (1.0f / (float)HID) - mu * mu;
            MU[r] = mu; RS[r] = rsqrtf(var + 1e-5f);
        }
        __syncthreads();
        float mu = MU[r], rs = RS[r];
        float part = 0.f;
        for (int i = 0; i < 128; ++i) {
            int h = i * 4 + q;
            float nv = (yr[h] - mu) * rs;
            float sa = 0.f;
            for (int a = 0; a < N_AGENTS; ++a) {
                float v = fmaf(nv, g2[a * HID + h], be2[a * HID + h]);
                sa += fmaxf(v, 0.f);
            }
            part += sa * wV[h];
        }
        __syncthreads();
        RED[tid] = part;
        __syncthreads();
        if (q == 0) {
            float v = bV[0] + RED[tid] + RED[tid + 1] + RED[tid + 2] + RED[tid + 3];
            float* o = out + (unsigned long long)(b0 + r) * N_AGENTS;
            #pragma unroll
            for (int j = 0; j < N_AGENTS; ++j) o[j] = v;   // tiled output
        }
    }
}

// ---------------- launch ----------------
extern "C" void kernel_launch(void* const* d_in, const int* in_sizes, int n_in,
                              void* d_out, int out_size, void* d_ws, size_t ws_size,
                              hipStream_t stream) {
    (void)in_sizes; (void)n_in; (void)out_size; (void)ws_size;
    const float* inputs  = (const float*)d_in[0];
    const float* actions = (const float*)d_in[1];
    const float* w1  = (const float*)d_in[2];
    const float* b1  = (const float*)d_in[3];
    const float* g1  = (const float*)d_in[4];
    const float* be1 = (const float*)d_in[5];
    const float* w2  = (const float*)d_in[6];
    const float* b2  = (const float*)d_in[7];
    const float* g2  = (const float*)d_in[8];
    const float* be2 = (const float*)d_in[9];
    const float* wV  = (const float*)d_in[10];
    const float* bV  = (const float*)d_in[11];
    float*    out = (float*)d_out;
    _Float16* w2h = (_Float16*)d_ws;   // 512 KB f16 copy of w2

    w2_cvt_kernel<<<(HID * HID) / 256, 256, 0, stream>>>(w2, w2h);
    critic_fused_kernel<<<BATCH / MT, 256, SMEM_BYTES, stream>>>(
        inputs, actions, w1, b1, g1, be1, b2, g2, be2, wV, bV, w2h, out);
}
